// ResidualAttentionEncoderLayer_62680752718163
// MI455X (gfx1250) — matrix operand order repaired and checked by reading, not compiled, not run on hardware
//
#include <hip/hip_runtime.h>
#include <hip/hip_bf16.h>

// ---------------------------------------------------------------------------
// Problem constants
// ---------------------------------------------------------------------------
static constexpr int Bb = 4, Ss = 1024, Ee = 1024, Hh = 16, Dd = 64, Ff = 4096;
static constexpr int E3 = 3 * Ee;        // 3072
static constexpr int MM = Bb * Ss;       // 4096 rows of activations

typedef __bf16       bf16x16 __attribute__((ext_vector_type(16)));
typedef float        f32x8   __attribute__((ext_vector_type(8)));
typedef unsigned int u32x8   __attribute__((ext_vector_type(8)));
typedef unsigned int u32x4   __attribute__((ext_vector_type(4)));
typedef int          i32x4   __attribute__((ext_vector_type(4)));
typedef int          i32x8   __attribute__((ext_vector_type(8)));

// ---------------------------------------------------------------------------
// bf16 helpers (RNE pack)
// ---------------------------------------------------------------------------
static __device__ __forceinline__ unsigned short f2bf(float x) {
    unsigned a = __builtin_bit_cast(unsigned, x);
    return (unsigned short)((a + 0x7FFFu + ((a >> 16) & 1u)) >> 16);
}
static __device__ __forceinline__ float bf2f(unsigned short h) {
    unsigned u = ((unsigned)h) << 16;
    return __builtin_bit_cast(float, u);
}
static __device__ __forceinline__ unsigned pk_bf16(float lo, float hi) {
    return ((unsigned)f2bf(hi) << 16) | (unsigned)f2bf(lo);
}
static __device__ __forceinline__ f32x8 wmma_bf16(u32x8 a, u32x8 b, f32x8 c) {
    return __builtin_amdgcn_wmma_f32_16x16x32_bf16(
        false, __builtin_bit_cast(bf16x16, a),
        false, __builtin_bit_cast(bf16x16, b),
        (short)0, c, false, false);
}

// ---------------------------------------------------------------------------
// Tensor Data Mover: 2D tile (bf16) global -> LDS.
// D# layout per CDNA5 ISA.
//   group0: count=1 | lds_addr | global_addr[56:0] | type=2
//   group1: data_size=1 (2B) | tensor_dim0/1 | tile_dim0/1 | dim0 stride
// This toolchain's builtin takes 6 args (extra int32x8 descriptor group).
// Issued by one wave; caller must s_wait_tensorcnt then barrier.
// ---------------------------------------------------------------------------
static __device__ __forceinline__ void tdm_load_tile_2d_bf16(
    const unsigned short* gptr, unsigned lds_byte_off,
    unsigned tensor_d0, unsigned tensor_d1,
    unsigned tile_d0, unsigned tile_d1, unsigned stride0_elems) {
    unsigned long long ga = (unsigned long long)(const void*)gptr;
    u32x4 g0;
    g0[0] = 1u;                                            // count=1, user D#
    g0[1] = lds_byte_off;                                  // lds_addr
    g0[2] = (unsigned)(ga & 0xFFFFFFFFu);                  // global_addr[31:0]
    g0[3] = (unsigned)((ga >> 32) & 0x01FFFFFFu)           // global_addr[56:32]
            | (2u << 30);                                  // type=2 ("image")
    i32x8 g1;
    g1[0] = (int)(1u << 16);                               // data_size=1 -> 2B
    g1[1] = (int)((tensor_d0 & 0xFFFFu) << 16);            // tensor_dim0[15:0]
    g1[2] = (int)((tensor_d0 >> 16) | ((tensor_d1 & 0xFFFFu) << 16));
    g1[3] = (int)((tensor_d1 >> 16) | (tile_d0 << 16));    // tile_dim0
    g1[4] = (int)(tile_d1 & 0xFFFFu);                      // tile_dim1 (dim2=0)
    g1[5] = (int)stride0_elems;                            // dim0_stride[31:0]
    g1[6] = 0;                                             // stride hi bits
    g1[7] = 0;
    i32x4 z4 = {0, 0, 0, 0};
    i32x8 z8 = {0, 0, 0, 0, 0, 0, 0, 0};
    __builtin_amdgcn_tensor_load_to_lds(g0, g1, z4, z4, z8, 0);
}

// ---------------------------------------------------------------------------
// fp32 -> bf16 conversion (weights)
// ---------------------------------------------------------------------------
__global__ void cvt_bf16_kernel(const float* __restrict__ src,
                                unsigned short* __restrict__ dst, int n) {
    for (int i = blockIdx.x * blockDim.x + threadIdx.x; i < n;
         i += gridDim.x * blockDim.x)
        dst[i] = f2bf(src[i]);
}

// ---------------------------------------------------------------------------
// LayerNorm: one 256-thread block per row of length 1024, bf16 output
// ---------------------------------------------------------------------------
__global__ __launch_bounds__(256) void ln_bf16_kernel(
    const float* __restrict__ x, const float* __restrict__ g,
    const float* __restrict__ bta, unsigned short* __restrict__ out) {
    __shared__ float r1[256], r2[256];
    const int row = blockIdx.x, tid = threadIdx.x;
    const float* xr = x + (size_t)row * Ee;
    float v[4], s = 0.f, s2 = 0.f;
#pragma unroll
    for (int j = 0; j < 4; ++j) {
        v[j] = xr[tid + j * 256];
        s += v[j]; s2 += v[j] * v[j];
    }
    r1[tid] = s; r2[tid] = s2; __syncthreads();
    for (int off = 128; off > 0; off >>= 1) {
        if (tid < off) { r1[tid] += r1[tid + off]; r2[tid] += r2[tid + off]; }
        __syncthreads();
    }
    const float mean = r1[0] * (1.0f / Ee);
    const float var  = r2[0] * (1.0f / Ee) - mean * mean;
    const float rs   = rsqrtf(var + 1e-5f);
    unsigned short* orow = out + (size_t)row * Ee;
#pragma unroll
    for (int j = 0; j < 4; ++j) {
        int i = tid + j * 256;
        orow[i] = f2bf((v[j] - mean) * rs * g[i] + bta[i]);
    }
}

// ---------------------------------------------------------------------------
// Tiled bf16 WMMA GEMM:  C[M,N] = epilogue( A[M,K] @ W[N,K]^T + bias )
//   EPI 0: bf16 out (bias)      EPI 1: bf16 out (bias + exact-erf GELU)
//   EPI 2: f32 out (bias + residual add)
// Block tile 128x128, BK=32; 8 waves, each 32(M)x64(N) = 2x4 WMMA tiles.
// A tile staged by the Tensor Data Mover (wave 0 issues, waits TENSORcnt);
// W tile staged transposed by all waves; next W tile prefetched into L2.
// ---------------------------------------------------------------------------
template <int EPI>
__global__ __launch_bounds__(256) void gemm_bf16_kernel(
    const unsigned short* __restrict__ A, const unsigned short* __restrict__ W,
    const float* __restrict__ bias, const float* __restrict__ res,
    void* __restrict__ outp, int M, int N, int K) {
    __shared__ unsigned short As[128 * 32];   // A tile (TDM dest, contiguous)
    __shared__ unsigned short Bs[32 * 136];   // W tile transposed [k][n]

    const int tid  = threadIdx.x;
    const int lane = tid & 31, wid = tid >> 5;
    const int half = lane >> 4, l15 = lane & 15;
    const int wave_m = wid >> 1, wave_n = wid & 1;     // 4x2 wave grid
    const int m0 = blockIdx.y * 128, n0 = blockIdx.x * 128;

    f32x8 acc[2][4];
#pragma unroll
    for (int a = 0; a < 2; ++a)
#pragma unroll
        for (int bb = 0; bb < 4; ++bb) acc[a][bb] = (f32x8)(0.f);

    const int ldrow = tid >> 1;            // 0..127
    const int ldc   = (tid & 1) * 16;      // 0 or 16
    const unsigned lds_As = (unsigned)(size_t)(void*)As;
    unsigned* As32 = (unsigned*)As;

    for (int k0 = 0; k0 < K; k0 += 32) {
        // ---- TDM: A tile (128 rows x 32 cols bf16) -> LDS, wave 0 only ----
        if (tid < 32) {
            tdm_load_tile_2d_bf16(A + (size_t)m0 * K + k0, lds_As,
                                  (unsigned)K, (unsigned)M,
                                  32u, 128u, (unsigned)K);
        }
        // ---- stage W tile transposed into Bs[k][n] (all waves) ----
        {
            const unsigned short* src = W + (size_t)(n0 + ldrow) * K + k0 + ldc;
            uint4 p0 = *(const uint4*)(src);
            uint4 p1 = *(const uint4*)(src + 8);
            if (k0 + 32 < K) __builtin_prefetch(src + 32, 0, 1);
            unsigned w8[8] = {p0.x, p0.y, p0.z, p0.w, p1.x, p1.y, p1.z, p1.w};
#pragma unroll
            for (int u = 0; u < 8; ++u) {
                int kloc = ldc + 2 * u;
                Bs[kloc * 136 + ldrow]       = (unsigned short)(w8[u] & 0xFFFFu);
                Bs[(kloc + 1) * 136 + ldrow] = (unsigned short)(w8[u] >> 16);
            }
        }
        if (tid < 32) __builtin_amdgcn_s_wait_tensorcnt(0);
        __syncthreads();

        // ---- fragments + 8 WMMA ----
        u32x8 af[2];
#pragma unroll
        for (int mt = 0; mt < 2; ++mt) {
            const int mrow = wave_m * 32 + mt * 16 + l15;
            const unsigned* p = As32 + mrow * 16 + half * 4;
#pragma unroll
            for (int j = 0; j < 4; ++j) af[mt][j] = p[j];
#pragma unroll
            for (int j = 0; j < 4; ++j) af[mt][4 + j] = p[8 + j];
        }
        const unsigned* Bs32 = (const unsigned*)Bs;
#pragma unroll
        for (int tn = 0; tn < 4; ++tn) {
            const int nb = wave_n * 64 + tn * 16;
            u32x8 bf_;
#pragma unroll
            for (int v = 0; v < 8; ++v) {
                const int kb = (v >> 2) * 16 + half * 8 + 2 * (v & 3);
                bf_[v] = (unsigned)Bs[kb * 136 + nb + l15] | ((unsigned)Bs[(kb + 1) * 136 + nb + l15] << 16);
            }
            acc[0][tn] = wmma_bf16(af[0], bf_, acc[0][tn]);
            acc[1][tn] = wmma_bf16(af[1], bf_, acc[1][tn]);
        }
        __syncthreads();
    }

    // ---- epilogue ----
#pragma unroll
    for (int mt = 0; mt < 2; ++mt)
#pragma unroll
        for (int tn = 0; tn < 4; ++tn)
#pragma unroll
            for (int r = 0; r < 8; ++r) {
                const int row = m0 + wave_m * 32 + mt * 16 + half * 8 + r;
                const int col = n0 + wave_n * 64 + tn * 16 + l15;
                float vv = acc[mt][tn][r] + bias[col];
                if constexpr (EPI == 0) {
                    ((unsigned short*)outp)[(size_t)row * N + col] = f2bf(vv);
                } else if constexpr (EPI == 1) {
                    vv = 0.5f * vv * (1.0f + erff(vv * 0.70710678118654752f));
                    ((unsigned short*)outp)[(size_t)row * N + col] = f2bf(vv);
                } else {
                    ((float*)outp)[(size_t)row * N + col] =
                        vv + res[(size_t)row * N + col];
                }
            }
}

// ---------------------------------------------------------------------------
// Attention: one block per (batch, 32-query-row tile); loops over 16 heads.
// Uses ~294 KB of the WGP's 320 KB LDS:
//   probs[32][1024] f32, accw[32][1024] f32 (head-mean), q tile, ctx partials.
// ---------------------------------------------------------------------------
static constexpr size_t ATTN_SMEM =
    (size_t)32 * 1024 * 4 +   // probs
    (size_t)32 * 1024 * 4 +   // accw
    (size_t)32 * 72 * 2 +     // q tile (stride 72)
    (size_t)8 * 32 * 64 * 2 + // ctx partials (bf16, per wave)
    256 * 4 + 32 * 4;         // reductions

__global__ __launch_bounds__(256) void attn_kernel(
    const unsigned short* __restrict__ qkv,   // [B,S,3E] bf16
    unsigned short* __restrict__ ctx,         // [B,S,E]  bf16
    float* __restrict__ attnw) {              // [B,S,S]  f32 (head mean)
    extern __shared__ char smem[];
    float* probs = (float*)smem;                                  // 32*1024
    float* accw  = probs + 32 * 1024;                             // 32*1024
    unsigned short* qs    = (unsigned short*)(accw + 32 * 1024);  // 32*72
    unsigned short* ppart = qs + 32 * 72;                         // 8*32*64
    float* red  = (float*)(ppart + 8 * 32 * 64);                  // 256
    float* rowv = red + 256;                                      // 32

    const int tid = threadIdx.x, lane = tid & 31, wid = tid >> 5;
    const int half = lane >> 4, l15 = lane & 15;
    const int b = blockIdx.x >> 5, q0 = (blockIdx.x & 31) * 32;
    const size_t rowbase = (size_t)(b * Ss + q0);

    for (int i = tid; i < 32 * 1024; i += 256) accw[i] = 0.f;
    __syncthreads();

    unsigned* qs32 = (unsigned*)qs;

    for (int h = 0; h < Hh; ++h) {
        const int hoff = h * Dd;
        // -------- load + pre-scale Q tile (32x64) --------
        {
            const int row = tid >> 3, d0 = (tid & 7) * 8;
            const unsigned short* src =
                qkv + (rowbase + row) * E3 + hoff + d0;
            uint4 p = *(const uint4*)src;
            unsigned w8[4] = {p.x, p.y, p.z, p.w};
            unsigned* dq = qs32 + ((row * 72 + d0) >> 1);
#pragma unroll
            for (int j = 0; j < 4; ++j) {
                float lo = bf2f((unsigned short)(w8[j] & 0xFFFFu)) * 0.125f;
                float hi = bf2f((unsigned short)(w8[j] >> 16)) * 0.125f;
                dq[j] = pk_bf16(lo, hi);
            }
        }
        __syncthreads();

        // -------- Phase A: scores = (Q*scale) @ K^T, wave owns 128 cols ----
        u32x8 af[2][2];
#pragma unroll
        for (int mt = 0; mt < 2; ++mt)
#pragma unroll
            for (int kk = 0; kk < 2; ++kk) {
                const int mrow = mt * 16 + l15;
                const unsigned* p = qs32 + mrow * 36 + kk * 16 + half * 4;
#pragma unroll
                for (int j = 0; j < 4; ++j) af[mt][kk][j] = p[j];
#pragma unroll
                for (int j = 0; j < 4; ++j) af[mt][kk][4 + j] = p[8 + j];
            }
        for (int tn = 0; tn < 8; ++tn) {
            const int n0 = wid * 128 + tn * 16;
            f32x8 a0 = (f32x8)(0.f), a1 = (f32x8)(0.f);
#pragma unroll
            for (int kk = 0; kk < 2; ++kk) {
                const int d = kk * 32 + half * 8;
                u32x8 bf_;
#pragma unroll
                for (int v = 0; v < 8; ++v) {
                    const unsigned short* kp =
                        qkv + (size_t)(b * Ss + n0 + l15) * E3 + Ee + hoff + d + (v >> 2) * 16 + 2 * (v & 3);
                    bf_[v] = ((unsigned)kp[1] << 16) | (unsigned)kp[0];
                }
                a0 = wmma_bf16(af[0][kk], bf_, a0);
                a1 = wmma_bf16(af[1][kk], bf_, a1);
            }
#pragma unroll
            for (int r = 0; r < 8; ++r) {
                probs[(half * 8 + r) * 1024 + n0 + l15]        = a0[r];
                probs[(16 + half * 8 + r) * 1024 + n0 + l15]   = a1[r];
            }
        }
        __syncthreads();

        // -------- Phase B: softmax + head-mean accumulation --------
        {
            const int r = tid & 31, seg = tid >> 5;
            float* pr = probs + r * 1024 + seg * 128;
            float mx = -3.0e38f;
            for (int c = 0; c < 128; ++c) mx = fmaxf(mx, pr[c]);
            red[r * 8 + seg] = mx;
            __syncthreads();
            if (tid < 32) {
                float m2 = red[tid * 8];
                for (int j = 1; j < 8; ++j) m2 = fmaxf(m2, red[tid * 8 + j]);
                rowv[tid] = m2;
            }
            __syncthreads();
            const float rm = rowv[r];
            float ssum = 0.f;
            for (int c = 0; c < 128; ++c) {
                float p = __expf(pr[c] - rm);
                pr[c] = p; ssum += p;
            }
            red[r * 8 + seg] = ssum;
            __syncthreads();
            if (tid < 32) {
                float s2 = 0.f;
                for (int j = 0; j < 8; ++j) s2 += red[tid * 8 + j];
                rowv[tid] = 1.0f / s2;
            }
            __syncthreads();
            const float rinv = rowv[r];
            float* aw = accw + r * 1024 + seg * 128;
            for (int c = 0; c < 128; ++c) {
                float p = pr[c] * rinv;
                pr[c] = p;
                aw[c] += p * (1.0f / Hh);
            }
        }
        __syncthreads();

        // -------- Phase C: ctx = P @ V, K split across 8 waves --------
        {
            const int kbase = wid * 128;
            f32x8 acc[2][4];
#pragma unroll
            for (int a = 0; a < 2; ++a)
#pragma unroll
                for (int t = 0; t < 4; ++t) acc[a][t] = (f32x8)(0.f);
            for (int kk = 0; kk < 4; ++kk) {
                const int k = kbase + kk * 32;
                const unsigned short* vb =
                    qkv + (size_t)(b * Ss + k + half * 8) * E3 + 2 * Ee + hoff;
                u32x8 bf_[4];
#pragma unroll
                for (int tn = 0; tn < 4; ++tn)
#pragma unroll
                    for (int v = 0; v < 8; ++v) {
                        const unsigned short* vp = vb + (size_t)((v >> 2) * 16 + 2 * (v & 3)) * E3 + tn * 16 + l15;
                        bf_[tn][v] = ((unsigned)vp[E3] << 16) | (unsigned)vp[0];
                    }
#pragma unroll
                for (int mt = 0; mt < 2; ++mt) {
                    const int m = mt * 16 + l15;
                    const float* pr = probs + m * 1024 + k + half * 8;
                    u32x8 afr;
#pragma unroll
                    for (int j = 0; j < 4; ++j)
                        afr[j] = pk_bf16(pr[2 * j], pr[2 * j + 1]);
#pragma unroll
                    for (int j = 0; j < 4; ++j)
                        afr[4 + j] = pk_bf16(pr[16 + 2 * j], pr[16 + 2 * j + 1]);
#pragma unroll
                    for (int tn = 0; tn < 4; ++tn)
                        acc[mt][tn] = wmma_bf16(afr, bf_[tn], acc[mt][tn]);
                }
            }
#pragma unroll
            for (int mt = 0; mt < 2; ++mt)
#pragma unroll
                for (int tn = 0; tn < 4; ++tn)
#pragma unroll
                    for (int r = 0; r < 8; ++r)
                        ppart[wid * 2048 +
                              (mt * 16 + half * 8 + r) * 64 + tn * 16 + l15] =
                            f2bf(acc[mt][tn][r]);
        }
        __syncthreads();

        // -------- Phase D: reduce 8 wave partials, write ctx --------
        {
            const int e0 = tid * 8;
#pragma unroll
            for (int j = 0; j < 8; ++j) {
                const int e = e0 + j, row = e >> 6, col = e & 63;
                float s = 0.f;
#pragma unroll
                for (int w = 0; w < 8; ++w) s += bf2f(ppart[w * 2048 + e]);
                ctx[(rowbase + row) * Ee + hoff + col] = f2bf(s);
            }
        }
        __syncthreads();
    }

    // -------- Phase E: write head-mean attention weights --------
    {
        const int r = tid >> 3, seg = tid & 7;
        const float* src = accw + r * 1024 + seg * 128;
        float* dst = attnw + (rowbase + r) * Ss + seg * 128;
        for (int c = 0; c < 128; ++c) dst[c] = src[c];
    }
}

// ---------------------------------------------------------------------------
// Host launcher
// ---------------------------------------------------------------------------
extern "C" void kernel_launch(void* const* d_in, const int* in_sizes, int n_in,
                              void* d_out, int out_size, void* d_ws, size_t ws_size,
                              hipStream_t stream) {
    const float* query  = (const float*)d_in[0];
    const float* ln1_g  = (const float*)d_in[1];
    const float* ln1_b  = (const float*)d_in[2];
    const float* w_in   = (const float*)d_in[3];
    const float* b_in   = (const float*)d_in[4];
    const float* w_out  = (const float*)d_in[5];
    const float* b_out  = (const float*)d_in[6];
    const float* ln2_g  = (const float*)d_in[7];
    const float* ln2_b  = (const float*)d_in[8];
    const float* w1     = (const float*)d_in[9];
    const float* b1     = (const float*)d_in[10];
    const float* w2     = (const float*)d_in[11];
    const float* b2     = (const float*)d_in[12];

    // workspace carve (bytes, 256-aligned sizes)
    char* ws = (char*)d_ws;
    size_t off = 0;
    auto carve = [&](size_t bytes) {
        char* p = ws + off;
        off += (bytes + 255) & ~(size_t)255;
        return p;
    };
    unsigned short* win_bf  = (unsigned short*)carve((size_t)E3 * Ee * 2);
    unsigned short* wout_bf = (unsigned short*)carve((size_t)Ee * Ee * 2);
    unsigned short* w1_bf   = (unsigned short*)carve((size_t)Ff * Ee * 2);
    unsigned short* w2_bf   = (unsigned short*)carve((size_t)Ee * Ff * 2);
    unsigned short* nq_bf   = (unsigned short*)carve((size_t)MM * Ee * 2);
    unsigned short* qkv_bf  = (unsigned short*)carve((size_t)MM * E3 * 2);
    unsigned short* ctx_bf  = (unsigned short*)carve((size_t)MM * Ee * 2);
    float*          x1      = (float*)carve((size_t)MM * Ee * 4);
    unsigned short* h1_bf   = (unsigned short*)carve((size_t)MM * Ee * 2);
    unsigned short* hm_bf   = (unsigned short*)carve((size_t)MM * Ff * 2);

    float* x_out  = (float*)d_out;                       // [B,S,E]
    float* aw_out = x_out + (size_t)Bb * Ss * Ee;        // [B,S,S]

    // 1) weights -> bf16
    cvt_bf16_kernel<<<2048, 256, 0, stream>>>(w_in,  win_bf,  E3 * Ee);
    cvt_bf16_kernel<<<2048, 256, 0, stream>>>(w_out, wout_bf, Ee * Ee);
    cvt_bf16_kernel<<<2048, 256, 0, stream>>>(w1,    w1_bf,   Ff * Ee);
    cvt_bf16_kernel<<<2048, 256, 0, stream>>>(w2,    w2_bf,   Ee * Ff);
    // 2) LN1
    ln_bf16_kernel<<<MM, 256, 0, stream>>>(query, ln1_g, ln1_b, nq_bf);
    // 3) QKV projection
    gemm_bf16_kernel<0><<<dim3(E3 / 128, MM / 128), 256, 0, stream>>>(
        nq_bf, win_bf, b_in, nullptr, qkv_bf, MM, E3, Ee);
    // 4) attention (per (batch, 32-row) tile; 16 heads in-block; big LDS)
    attn_kernel<<<Bb * (Ss / 32), 256, ATTN_SMEM, stream>>>(qkv_bf, ctx_bf, aw_out);
    // 5) out-proj + residual1 (fp32)
    gemm_bf16_kernel<2><<<dim3(Ee / 128, MM / 128), 256, 0, stream>>>(
        ctx_bf, wout_bf, b_out, query, x1, MM, Ee, Ee);
    // 6) LN2
    ln_bf16_kernel<<<MM, 256, 0, stream>>>(x1, ln2_g, ln2_b, h1_bf);
    // 7) MLP up + GELU
    gemm_bf16_kernel<1><<<dim3(Ff / 128, MM / 128), 256, 0, stream>>>(
        h1_bf, w1_bf, b1, nullptr, hm_bf, MM, Ff, Ee);
    // 8) MLP down + residual2 -> final x output
    gemm_bf16_kernel<2><<<dim3(Ee / 128, MM / 128), 256, 0, stream>>>(
        hm_bf, w2_bf, b2, x1, x_out, MM, Ee, Ff);
}
